// BaichuanFlashAttention2_4801773437259
// MI455X (gfx1250) — matrix operand order repaired and checked by reading, not compiled, not run on hardware
//
#include <hip/hip_runtime.h>

typedef __bf16 bf16_t;
typedef __attribute__((ext_vector_type(16))) __bf16 v16bf;
typedef __attribute__((ext_vector_type(8)))  float  v8f;
typedef __attribute__((ext_vector_type(4)))  unsigned int u32x4;
typedef __attribute__((ext_vector_type(4)))  int i32x4;
typedef __attribute__((ext_vector_type(8)))  int i32x8;

#define S_LEN 2048
#define HIDN  4096
#define NH    32
#define NKV   8
#define HD    128
#define QKVN  6144   // H*D + 2*KV*D

#if defined(__HIP_DEVICE_COMPILE__) && __has_builtin(__builtin_amdgcn_tensor_load_to_lds)
#define HAVE_TDM 1
#else
#define HAVE_TDM 0
#endif

union Frag {
    v16bf v;
    uint4 u[2];
    unsigned short h[16];
};

__device__ __forceinline__ unsigned short f2bf(float f) {
    union { float f; unsigned u; } x; x.f = f;
    unsigned r = x.u + 0x7fffu + ((x.u >> 16) & 1u);   // round-to-nearest-even
    return (unsigned short)(r >> 16);
}

#if HAVE_TDM
// 2-D TDM tile load: tile_d0 contiguous 2-byte elements x tile_d1 rows
// (row stride = stride0 elements) from gsrc into LDS at lds_off bytes.
// D# per CDNA5 ISA 8.3/8.4: count=1, type=2, data_size=1 (2B), groups 2/3
// zero => 2-D tensor.
__device__ __forceinline__ void tdm_load_2d(const unsigned short* gsrc,
                                            unsigned lds_off,
                                            unsigned tensor_d0, unsigned tensor_d1,
                                            unsigned stride0,
                                            unsigned tile_d0, unsigned tile_d1) {
    unsigned long long ga = (unsigned long long)(size_t)gsrc;
    u32x4 g0;
    g0[0] = 1u;                                          // count=1, user D#
    g0[1] = lds_off;                                     // lds_addr (bytes)
    g0[2] = (unsigned)(ga & 0xffffffffu);                // global_addr[31:0]
    g0[3] = (unsigned)((ga >> 32) & 0x1ffffffu)          // global_addr[56:32]
            | 0x80000000u;                               // type=2 (bits 127:126)
    i32x8 g1;
    g1[0] = (int)(1u << 16);                             // data_size=1 (2 bytes)
    g1[1] = (int)((tensor_d0 & 0xffffu) << 16);          // tensor_dim0[15:0]
    g1[2] = (int)((tensor_d0 >> 16) | ((tensor_d1 & 0xffffu) << 16));
    g1[3] = (int)((tensor_d1 >> 16) | (tile_d0 << 16));  // tile_dim0
    g1[4] = (int)(tile_d1 & 0xffffu);                    // tile_dim1, tile_dim2=0
    g1[5] = (int)stride0;                                // tensor_dim0_stride lo
    g1[6] = 0;
    g1[7] = 0;
    i32x4 z4 = {0, 0, 0, 0};
#if defined(__clang_major__) && __clang_major__ >= 23
    i32x8 z8 = {0, 0, 0, 0, 0, 0, 0, 0};
    __builtin_amdgcn_tensor_load_to_lds(g0, g1, z4, z4, z8, 0);
#else
    __builtin_amdgcn_tensor_load_to_lds(g0, g1, z4, z4, 0);
#endif
}

__device__ __forceinline__ void tdm_wait() {
#if __has_builtin(__builtin_amdgcn_s_wait_tensorcnt)
    __builtin_amdgcn_s_wait_tensorcnt(0);
#else
    asm volatile("s_wait_tensorcnt 0x0" ::: "memory");
#endif
}
#endif // HAVE_TDM

// ---------------- fp32 -> bf16 convert ----------------
__global__ void cvt_bf16_kernel(const float* __restrict__ in,
                                unsigned short* __restrict__ out, int n) {
    int i = blockIdx.x * blockDim.x + threadIdx.x;
    if (i < n) out[i] = f2bf(in[i]);
}

// ---------------- NT GEMM: C[M,N] = A[M,K] * B[N,K]^T, bf16 in, f32 out ----
// 256 threads = 8 waves. Block tile 128x128 (wave tile 16x128).
// B tile (128 rows x 32 k, 8 KB) staged in LDS: TDM async copy when
// available (double-buffered, overlapped with WMMA), else cooperative loads.
__global__ __launch_bounds__(256)
void gemm_nt_bf16_kernel(const unsigned short* __restrict__ A,
                         const unsigned short* __restrict__ B,
                         float* __restrict__ C, int M, int N, int K) {
    __shared__ __align__(128) unsigned short Btile[2][128][32];
    const int tid   = threadIdx.x;
    const int lane  = tid & 31;
    const int wave  = tid >> 5;
    const int lhalf = lane >> 4;
    const int l16   = lane & 15;
    const int m0 = blockIdx.x * 128 + wave * 16;
    const int n0 = blockIdx.y * 128;

    v8f vzero = {};
    v8f acc[8];
#pragma unroll
    for (int t = 0; t < 8; ++t) acc[t] = vzero;

    const int nsteps = K >> 5;

    // stage k-step 0 into buffer 0
#if HAVE_TDM
    if (wave == 0)
        tdm_load_2d(B + (size_t)n0 * K, (unsigned)(size_t)&Btile[0][0][0],
                    (unsigned)K, (unsigned)N, (unsigned)K, 32u, 128u);
#else
    {
        const int r = tid >> 1, c = (tid & 1) * 16;
        const unsigned short* g = B + (size_t)(n0 + r) * K + c;
        uint4 a0 = *(const uint4*)g;
        uint4 a1 = *(const uint4*)(g + 8);
        *(uint4*)&Btile[0][r][c]     = a0;
        *(uint4*)&Btile[0][r][c + 8] = a1;
    }
#endif

    for (int i = 0; i < nsteps; ++i) {
        const int k0 = i << 5;
#if HAVE_TDM
        if (wave == 0) tdm_wait();
#endif
        __syncthreads();   // buffer i&1 ready for all waves; prior reads done
        if (i + 1 < nsteps) {
#if HAVE_TDM
            if (wave == 0)
                tdm_load_2d(B + (size_t)n0 * K + k0 + 32,
                            (unsigned)(size_t)&Btile[(i + 1) & 1][0][0],
                            (unsigned)K, (unsigned)N, (unsigned)K, 32u, 128u);
#else
            const int r = tid >> 1, c = (tid & 1) * 16;
            const unsigned short* g = B + (size_t)(n0 + r) * K + k0 + 32 + c;
            uint4 a0 = *(const uint4*)g;
            uint4 a1 = *(const uint4*)(g + 8);
            *(uint4*)&Btile[(i + 1) & 1][r][c]     = a0;
            *(uint4*)&Btile[(i + 1) & 1][r][c + 8] = a1;
#endif
        }

        // A fragment (contiguous: k = 16*lhalf + s, same order as B)
        Frag a;
        const unsigned short* pa = A + (size_t)(m0 + l16) * K + k0 + lhalf * 16;
        a.u[0] = *(const uint4*)pa;
        a.u[1] = *(const uint4*)(pa + 8);

        // preload all 8 B fragments, then burst the WMMAs
        const unsigned short* bt = &Btile[i & 1][0][0];
        Frag b[8];
#pragma unroll
        for (int t = 0; t < 8; ++t)
            b[t].v = *(const v16bf*)(bt + (t * 16 + l16) * 32 + lhalf * 16);
#pragma unroll
        for (int t = 0; t < 8; ++t)
            acc[t] = __builtin_amdgcn_wmma_f32_16x16x32_bf16(
                false, a.v, false, b[t].v, (short)0, acc[t], false, false);
    }
#pragma unroll
    for (int t = 0; t < 8; ++t)
#pragma unroll
        for (int j = 0; j < 8; ++j)
            C[(size_t)(m0 + 8 * lhalf + j) * N + n0 + t * 16 + l16] = acc[t][j];
}

// ---------------- RoPE on Q ----------------
__global__ void rope_q_kernel(const float* __restrict__ qkv,
                              unsigned short* __restrict__ Qh) {
    int idx = blockIdx.x * blockDim.x + threadIdx.x;
    if (idx >= S_LEN * NH * 64) return;
    int i = idx & 63;
    int h = (idx >> 6) & 31;
    int s = idx >> 11;
    const float* base = qkv + (size_t)s * QKVN + h * HD;
    float x1 = base[i], x2 = base[i + 64];
    float inv = __powf(100000.0f, -(float)i * (1.0f / 64.0f));
    float ang = (float)s * inv;
    float sn, cs; __sincosf(ang, &sn, &cs);
    unsigned short* ob = Qh + ((size_t)h * S_LEN + s) * HD;
    ob[i]      = f2bf(x1 * cs - x2 * sn);
    ob[i + 64] = f2bf(x1 * sn + x2 * cs);
}

// ---------------- causal conv(2) + RoPE on K ----------------
__global__ void convrope_k_kernel(const float* __restrict__ qkv,
                                  const float* __restrict__ ck,
                                  unsigned short* __restrict__ Kh) {
    int idx = blockIdx.x * blockDim.x + threadIdx.x;
    if (idx >= S_LEN * NKV * 64) return;
    int i  = idx & 63;
    int kv = (idx >> 6) & 7;
    int s  = idx >> 9;
    size_t off = (size_t)s * QKVN + NH * HD + kv * HD;
    float c0 = ck[kv * 2], c1 = ck[kv * 2 + 1];
    float x1 = c1 * qkv[off + i];
    float x2 = c1 * qkv[off + i + 64];
    if (s > 0) {
        x1 += c0 * qkv[off - QKVN + i];
        x2 += c0 * qkv[off - QKVN + i + 64];
    }
    float inv = __powf(100000.0f, -(float)i * (1.0f / 64.0f));
    float ang = (float)s * inv;
    float sn, cs; __sincosf(ang, &sn, &cs);
    unsigned short* ob = Kh + ((size_t)kv * S_LEN + s) * HD;
    ob[i]      = f2bf(x1 * cs - x2 * sn);
    ob[i + 64] = f2bf(x1 * sn + x2 * cs);
}

// ---------------- causal conv(2) on V + transpose to [kv][d][s] ----------
__global__ void conv_vt_kernel(const float* __restrict__ qkv,
                               const float* __restrict__ cv,
                               unsigned short* __restrict__ Vt) {
    int idx = blockIdx.x * blockDim.x + threadIdx.x;
    if (idx >= NKV * HD * S_LEN) return;
    int s  = idx & 2047;
    int d  = (idx >> 11) & 127;
    int kv = idx >> 18;
    size_t off = (size_t)s * QKVN + (NH * HD + NKV * HD) + kv * HD + d;
    float x = cv[kv * 2 + 1] * qkv[off];
    if (s > 0) x += cv[kv * 2] * qkv[off - QKVN];
    Vt[(size_t)kv * HD * S_LEN + (size_t)d * S_LEN + s] = f2bf(x);
}

// ---------------- flash attention ----------------
// Block = 4 waves = 4 consecutive 16-row q-tiles of the SAME head.
// K/V tiles (8 KB each) staged once per block in LDS, double-buffered via
// TDM (wave 0 issues, TENSORcnt + barrier publishes), shared by all 4 waves.
__global__ __launch_bounds__(128)
void flash_attn_kernel(const unsigned short* __restrict__ Qh,
                       const unsigned short* __restrict__ Kh,
                       const unsigned short* __restrict__ Vt,
                       unsigned short* __restrict__ Oattn) {
    __shared__ __align__(128) unsigned short Kt[2][32][128];  // kv rows x d
    __shared__ __align__(128) unsigned short Vb[2][128][32];  // d rows x kv
    __shared__ __align__(64)  unsigned short Pl[4][16][32];   // per-wave P tile

    const int tid   = threadIdx.x;
    const int lane  = tid & 31;
    const int wave  = tid >> 5;
    const int lhalf = lane >> 4;
    const int l16   = lane & 15;

    const int h      = blockIdx.x >> 5;      // 32 q-blocks of 64 rows per head
    const int blockQ = blockIdx.x & 31;
    const int q0     = blockQ * 64 + wave * 16;
    const int hkv    = h >> 2;               // H / KV = 4
    const float scale = 0.08838834764831845f; // 1/sqrt(128)

    // Q fragments: 4 d-slices of 32 (contiguous k = 16*lhalf + s, as K frags)
    Frag qf[4];
#pragma unroll
    for (int ds = 0; ds < 4; ++ds) {
        const unsigned short* pq =
            Qh + ((size_t)h * S_LEN + q0 + l16) * HD + ds * 32 + lhalf * 16;
        qf[ds].u[0] = *(const uint4*)pq;
        qf[ds].u[1] = *(const uint4*)(pq + 8);
    }

    // stage K/V tiles for kv block at `kb` into buffer `buf`
    auto stage = [&](int buf, int kb) {
#if HAVE_TDM
        if (wave == 0) {
            tdm_load_2d(Kh + ((size_t)hkv * S_LEN + kb) * HD,
                        (unsigned)(size_t)&Kt[buf][0][0],
                        128u, (1u << 24), (unsigned)HD, 128u, 32u);
            tdm_load_2d(Vt + (size_t)hkv * HD * S_LEN + kb,
                        (unsigned)(size_t)&Vb[buf][0][0],
                        (1u << 24), (1u << 24), (unsigned)S_LEN, 32u, 128u);
        }
#else
        {   // K: 32 rows x 256B ; 128 threads x 64B
            const int r = tid >> 2, c = (tid & 3) * 32;
            const uint4* g = (const uint4*)(Kh + ((size_t)hkv * S_LEN + kb + r) * HD + c);
            uint4* d = (uint4*)&Kt[buf][r][c];
#pragma unroll
            for (int q = 0; q < 4; ++q) d[q] = g[q];
        }
        {   // V: 128 rows x 64B ; one row per thread
            const uint4* g = (const uint4*)(Vt + ((size_t)(hkv * HD + tid)) * S_LEN + kb);
            uint4* d = (uint4*)&Vb[buf][tid][0];
#pragma unroll
            for (int q = 0; q < 4; ++q) d[q] = g[q];
        }
#endif
    };

    v8f vzero = {};
    v8f o[8];
#pragma unroll
    for (int t = 0; t < 8; ++t) o[t] = vzero;
    float mrow[8], lrow[8];
#pragma unroll
    for (int j = 0; j < 8; ++j) { mrow[j] = -1.0e30f; lrow[j] = 0.0f; }

    const int nkb_max = 2 * blockQ + 2;   // covers wave 3's causal range
    stage(0, 0);

    for (int bi = 0; bi < nkb_max; ++bi) {
        const int kb  = bi * 32;
        const int cur = bi & 1;
#if HAVE_TDM
        if (wave == 0) tdm_wait();
#endif
        __syncthreads();   // buffer `cur` ready; prior reads of other buf done
        if (bi + 1 < nkb_max) stage((bi + 1) & 1, kb + 32);

        if (kb > q0 + 15) continue;   // uniform per wave: fully-masked block

        // S = Q K^T for 32 kv columns (two 16-wide chunks) from LDS
        v8f s0 = vzero, s1 = vzero;
#pragma unroll
        for (int ds = 0; ds < 4; ++ds) {
            Frag kf;
            kf.v = *(const v16bf*)&Kt[cur][l16][ds * 32 + lhalf * 16];
            s0 = __builtin_amdgcn_wmma_f32_16x16x32_bf16(
                false, qf[ds].v, false, kf.v, (short)0, s0, false, false);
        }
#pragma unroll
        for (int ds = 0; ds < 4; ++ds) {
            Frag kf;
            kf.v = *(const v16bf*)&Kt[cur][16 + l16][ds * 32 + lhalf * 16];
            s1 = __builtin_amdgcn_wmma_f32_16x16x32_bf16(
                false, qf[ds].v, false, kf.v, (short)0, s1, false, false);
        }

        // scale + causal mask (rows: q0 + 8*lhalf + j ; cols: kb + c*16 + l16)
        const int rowb = q0 + 8 * lhalf;
        const bool need_mask = (kb + 31 > q0);
#pragma unroll
        for (int j = 0; j < 8; ++j) {
            float a0 = s0[j] * scale, a1 = s1[j] * scale;
            if (need_mask) {
                if (kb + l16      > rowb + j) a0 = -1.0e30f;
                if (kb + 16 + l16 > rowb + j) a1 = -1.0e30f;
            }
            s0[j] = a0; s1[j] = a1;
        }

        // row max across 16 lanes of each half-wave
        float mx[8];
#pragma unroll
        for (int j = 0; j < 8; ++j) mx[j] = fmaxf(s0[j], s1[j]);
#pragma unroll
        for (int msk = 1; msk <= 8; msk <<= 1)
#pragma unroll
            for (int j = 0; j < 8; ++j)
                mx[j] = fmaxf(mx[j], __shfl_xor(mx[j], msk));

        float alpha[8];
#pragma unroll
        for (int j = 0; j < 8; ++j) {
            float mn = fmaxf(mrow[j], mx[j]);
            alpha[j] = __expf(mrow[j] - mn);
            mrow[j]  = mn;
        }

        float p0[8], p1[8], rs[8];
#pragma unroll
        for (int j = 0; j < 8; ++j) {
            p0[j] = __expf(s0[j] - mrow[j]);
            p1[j] = __expf(s1[j] - mrow[j]);
            rs[j] = p0[j] + p1[j];
        }
#pragma unroll
        for (int msk = 1; msk <= 8; msk <<= 1)
#pragma unroll
            for (int j = 0; j < 8; ++j)
                rs[j] += __shfl_xor(rs[j], msk);
#pragma unroll
        for (int j = 0; j < 8; ++j) lrow[j] = lrow[j] * alpha[j] + rs[j];

        // rescale O accumulators
#pragma unroll
        for (int t = 0; t < 8; ++t)
#pragma unroll
            for (int j = 0; j < 8; ++j) o[t][j] *= alpha[j];

        // P: D-layout (row-in-VGPR) -> LDS -> A-layout (row-in-lane)
#pragma unroll
        for (int j = 0; j < 8; ++j) {
            Pl[wave][8 * lhalf + j][l16]      = f2bf(p0[j]);
            Pl[wave][8 * lhalf + j][16 + l16] = f2bf(p1[j]);
        }
        asm volatile("s_wait_dscnt 0x0" ::: "memory");
        Frag pf;
        const unsigned short* pp = &Pl[wave][l16][lhalf * 16];
        pf.u[0] = *(const uint4*)pp;
        pf.u[1] = *(const uint4*)(pp + 8);

        // O += P * V from LDS
#pragma unroll
        for (int dc = 0; dc < 8; ++dc) {
            Frag vf;
            vf.v = *(const v16bf*)&Vb[cur][dc * 16 + l16][lhalf * 16];
            o[dc] = __builtin_amdgcn_wmma_f32_16x16x32_bf16(
                false, pf.v, false, vf.v, (short)0, o[dc], false, false);
        }
    }

    // epilogue: scale by 1/l, store bf16 attn output [S][H*D]
    float rinv[8];
#pragma unroll
    for (int j = 0; j < 8; ++j) rinv[j] = 1.0f / lrow[j];
#pragma unroll
    for (int dc = 0; dc < 8; ++dc)
#pragma unroll
        for (int j = 0; j < 8; ++j)
            Oattn[(size_t)(q0 + 8 * lhalf + j) * HIDN + h * HD + dc * 16 + l16] =
                f2bf(o[dc][j] * rinv[j]);
}

// ---------------- host launch ----------------
extern "C" void kernel_launch(void* const* d_in, const int* in_sizes, int n_in,
                              void* d_out, int out_size, void* d_ws, size_t ws_size,
                              hipStream_t stream) {
    (void)in_sizes; (void)n_in; (void)out_size; (void)ws_size;
    const float* hidden = (const float*)d_in[0];
    const float* w_pack = (const float*)d_in[1];
    const float* w_o    = (const float*)d_in[2];
    const float* conv_k = (const float*)d_in[3];
    const float* conv_v = (const float*)d_in[4];
    float* out = (float*)d_out;

    char* ws = (char*)d_ws;
    unsigned short* hid_bf   = (unsigned short*)ws; ws += (size_t)S_LEN * HIDN * 2;
    unsigned short* wpack_bf = (unsigned short*)ws; ws += (size_t)QKVN * HIDN * 2;
    unsigned short* wo_bf    = (unsigned short*)ws; ws += (size_t)HIDN * HIDN * 2;
    float*          qkv      = (float*)ws;          ws += (size_t)S_LEN * QKVN * 4;
    unsigned short* Qh       = (unsigned short*)ws; ws += (size_t)NH  * S_LEN * HD * 2;
    unsigned short* Kh       = (unsigned short*)ws; ws += (size_t)NKV * S_LEN * HD * 2;
    unsigned short* Vt       = (unsigned short*)ws; ws += (size_t)NKV * S_LEN * HD * 2;
    unsigned short* attn     = (unsigned short*)ws; ws += (size_t)S_LEN * HIDN * 2;

    int n1 = S_LEN * HIDN;
    cvt_bf16_kernel<<<(n1 + 255) / 256, 256, 0, stream>>>(hidden, hid_bf, n1);
    int n2 = QKVN * HIDN;
    cvt_bf16_kernel<<<(n2 + 255) / 256, 256, 0, stream>>>(w_pack, wpack_bf, n2);
    int n3 = HIDN * HIDN;
    cvt_bf16_kernel<<<(n3 + 255) / 256, 256, 0, stream>>>(w_o, wo_bf, n3);

    // qkv = hidden @ w_pack^T
    gemm_nt_bf16_kernel<<<dim3(S_LEN / 128, QKVN / 128), 256, 0, stream>>>(
        hid_bf, wpack_bf, qkv, S_LEN, QKVN, HIDN);

    int nq = S_LEN * NH * 64;
    rope_q_kernel<<<(nq + 255) / 256, 256, 0, stream>>>(qkv, Qh);
    int nk = S_LEN * NKV * 64;
    convrope_k_kernel<<<(nk + 255) / 256, 256, 0, stream>>>(qkv, conv_k, Kh);
    int nv = NKV * HD * S_LEN;
    conv_vt_kernel<<<(nv + 255) / 256, 256, 0, stream>>>(qkv, conv_v, Vt);

    // attention: H heads x (S/64) 64-row q blocks, 4 waves per block
    flash_attn_kernel<<<dim3(NH * (S_LEN / 64)), 128, 0, stream>>>(
        Qh, Kh, Vt, attn);

    // out = attn @ w_o^T
    gemm_nt_bf16_kernel<<<dim3(S_LEN / 128, HIDN / 128), 256, 0, stream>>>(
        attn, wo_bf, out, S_LEN, HIDN, HIDN);
}
